// DCNv2_1838246003413
// MI455X (gfx1250) — matrix operand order repaired and checked by reading, not compiled, not run on hardware
//
#include <hip/hip_runtime.h>
#include <hip/hip_bf16.h>

// ---- CDNA5 WMMA vector types ----
typedef __attribute__((ext_vector_type(16))) _Float16 v16h;
typedef __attribute__((ext_vector_type(8)))  float    v8f;
typedef __attribute__((ext_vector_type(4)))  int      v4i;
typedef __attribute__((ext_vector_type(4)))  _Float16 v4h;

#define BB   4
#define CC   256
#define HH   64
#define WWd  64
#define OO   256
#define KT   9
#define HWp  (HH * WWd)      // 4096
#define NPIX (BB * HWp)      // 16384
#define KCt  (CC * KT)       // 2304 (GEMM K dim, tap-major: kc = k*C + c)

// ---------------------------------------------------------------------------
// Kernel 1: x [B,C,H,W] f32  ->  xt [B,H*W,C] f32 (NHWC) via LDS tile transpose
// ---------------------------------------------------------------------------
__global__ __launch_bounds__(256) void k_nchw2nhwc(const float* __restrict__ x,
                                                   float* __restrict__ xt) {
  __shared__ float tile[32][33];
  int blk = blockIdx.x;          // 4 * 8 * 128 = 4096 blocks
  int pb  = blk & 127;           // 32-pixel chunk of flattened HW
  int cb  = (blk >> 7) & 7;      // 32-channel chunk
  int b   = blk >> 10;
  int j = threadIdx.x & 31;
  int i = threadIdx.x >> 5;
#pragma unroll
  for (int r = 0; r < 4; ++r) {
    int c = i + r * 8;
    tile[c][j] = x[((size_t)(b * CC + cb * 32 + c)) * HWp + pb * 32 + j];
  }
  __syncthreads();
#pragma unroll
  for (int r = 0; r < 4; ++r) {
    int p = i + r * 8;
    xt[((size_t)(b * HWp + pb * 32 + p)) * CC + cb * 32 + j] = tile[j][p];
  }
}

// ---------------------------------------------------------------------------
// Kernel 2: weight [O,C,3,3] f32 -> wrT [O][k*C + c] f16 (row-major, K'-contig)
// ---------------------------------------------------------------------------
__global__ __launch_bounds__(256) void k_wreorder(const float* __restrict__ w,
                                                  _Float16* __restrict__ wrT) {
  int tid = blockIdx.x * 256 + threadIdx.x;
  if (tid >= OO * KCt) return;
  int o = tid / KCt;
  int r = tid - o * KCt;
  int k = r >> 8;        // r / 256
  int c = r & 255;
  wrT[tid] = (_Float16)w[(size_t)o * KCt + c * KT + k];
}

// ---------------------------------------------------------------------------
// Kernel 3: per (tap, pixel): bilinear weights (mask+validity folded) + 4
// clamped NHWC corner BYTE offsets. Layout tap-major: idx = k*NPIX + n.
// ---------------------------------------------------------------------------
__global__ __launch_bounds__(256) void k_samp(const float* __restrict__ off,
                                              const float* __restrict__ msk,
                                              float4* __restrict__ sw,
                                              uint4* __restrict__ so) {
  int tid = blockIdx.x * 256 + threadIdx.x;
  if (tid >= KT * NPIX) return;
  int k = tid / NPIX;
  int n = tid - k * NPIX;
  int b   = n >> 12;
  int pix = n & (HWp - 1);
  int ho = pix >> 6, wo = pix & 63;

  float oy = off[((size_t)(b * 2 * KT + 2 * k)) * HWp + pix];
  float ox = off[((size_t)(b * 2 * KT + 2 * k + 1)) * HWp + pix];
  float m  = msk[((size_t)(b * KT + k)) * HWp + pix];

  float py = oy + (float)(ho - 1 + (k / 3));   // STRIDE=1, PAD=1, DIL=1
  float px = ox + (float)(wo - 1 + (k % 3));
  float fy = floorf(py), fx = floorf(px);
  float ly = py - fy, lx = px - fx;
  float hy = 1.f - ly, hx = 1.f - lx;
  int y0 = (int)fy, x0 = (int)fx;
  int y1 = y0 + 1,  x1 = x0 + 1;
  bool vy0 = (y0 >= 0) && (y0 < HH);
  bool vy1 = (y1 >= 0) && (y1 < HH);
  bool vx0 = (x0 >= 0) && (x0 < WWd);
  bool vx1 = (x1 >= 0) && (x1 < WWd);
  float w00 = (vy0 && vx0) ? hy * hx * m : 0.f;
  float w01 = (vy0 && vx1) ? hy * lx * m : 0.f;
  float w10 = (vy1 && vx0) ? ly * hx * m : 0.f;
  float w11 = (vy1 && vx1) ? ly * lx * m : 0.f;
  int y0c = y0 < 0 ? 0 : (y0 > HH - 1 ? HH - 1 : y0);
  int y1c = y1 < 0 ? 0 : (y1 > HH - 1 ? HH - 1 : y1);
  int x0c = x0 < 0 ? 0 : (x0 > WWd - 1 ? WWd - 1 : x0);
  int x1c = x1 < 0 ? 0 : (x1 > WWd - 1 ? WWd - 1 : x1);
  unsigned base = (unsigned)(b * HWp);
  uint4 o4;  // byte offsets into xt (max ~16.8 MB, fits u32)
  o4.x = (base + y0c * WWd + x0c) * (CC * 4u);
  o4.y = (base + y0c * WWd + x1c) * (CC * 4u);
  o4.z = (base + y1c * WWd + x0c) * (CC * 4u);
  o4.w = (base + y1c * WWd + x1c) * (CC * 4u);
  sw[tid] = make_float4(w00, w01, w10, w11);
  so[tid] = o4;
}

// ---------------------------------------------------------------------------
// Kernel 4: fused bilinear-im2col + f16 WMMA GEMM.
//   D[o, n] = sum_{k,c} wrT[o][k*C+c] * col[k,c][n]     (f32 accumulate)
// Block: 256 thr (8 waves, 2x4), tile 128(M) x 128(N), K-step = 32 channels.
// Double-buffered column tile -> exactly one barrier per K-step.
// ---------------------------------------------------------------------------
union FragAB { v16h h; v4i i2[2]; };

__global__ __launch_bounds__(256) void k_dcn_gemm(
    const _Float16* __restrict__ wrT, const float* __restrict__ xt,
    const float4* __restrict__ sw, const uint4* __restrict__ so,
    const float* __restrict__ bias, float* __restrict__ out) {
  __shared__ __align__(16) _Float16 colT[2][128 * 40]; // [buf][n][c], 40-half pitch
  __shared__ float4 sW[KT * 128];                      // all taps for this n-tile
  __shared__ uint4  sO[KT * 128];

  const int t    = threadIdx.x;
  const int lane = t & 31;
  const int wave = t >> 5;
  const int wm   = wave >> 2;        // 0..1 : 64 M-rows each
  const int wn   = wave & 3;         // 0..3 : 32 N-cols each
  const int l16  = lane & 15;
  const int lh   = lane >> 4;

  const int mblk = blockIdx.x & 1;
  const int nblk = blockIdx.x >> 1;
  const int n0 = nblk * 128;
  const int m0 = mblk * 128;

  const int c4   = (t & 7) * 4;      // 4 consecutive channels per thread
  const int ngrp = t >> 3;           // 32 n-groups; each covers n, n+32, n+64, n+96

  // Preload the whole sampling table for this pixel tile (9 taps x 128 px).
  for (int idx = t; idx < KT * 128; idx += 256) {
    int kk = idx >> 7, nl = idx & 127;
    sW[idx] = sw[kk * NPIX + n0 + nl];
    sO[idx] = so[kk * NPIX + n0 + nl];
  }
  __syncthreads();

  v8f acc[4][2] = {};
  const char* xtb  = (const char*)xt;
  const char* wrTb = (const char*)wrT;

  for (int step = 0; step < KT * 8; ++step) {
    const int k  = step >> 3;
    const int cb = step & 7;
    const int p  = step & 1;
    // ---- build 128 x 32 f16 column tile (bilinear gather, L2-resident) ----
    const unsigned chb = (unsigned)((cb * 32 + c4) * 4);   // channel byte offset
#pragma unroll
    for (int i = 0; i < 4; ++i) {
      int nl = ngrp + 32 * i;
      float4 wv = sW[k * 128 + nl];
      uint4  ov = sO[k * 128 + nl];
      float4 r00 = *(const float4*)(xtb + (ov.x + chb));
      float4 r01 = *(const float4*)(xtb + (ov.y + chb));
      float4 r10 = *(const float4*)(xtb + (ov.z + chb));
      float4 r11 = *(const float4*)(xtb + (ov.w + chb));
      float a0 = wv.x * r00.x + wv.y * r01.x + wv.z * r10.x + wv.w * r11.x;
      float a1 = wv.x * r00.y + wv.y * r01.y + wv.z * r10.y + wv.w * r11.y;
      float a2 = wv.x * r00.z + wv.y * r01.z + wv.z * r10.z + wv.w * r11.z;
      float a3 = wv.x * r00.w + wv.y * r01.w + wv.z * r10.w + wv.w * r11.w;
      v4h hv = {(_Float16)a0, (_Float16)a1, (_Float16)a2, (_Float16)a3};
      *(v4h*)(&colT[p][nl * 40 + c4]) = hv;
    }
    __syncthreads();   // the only barrier in the K-step

    // ---- A fragments straight from L2-resident reordered weight ----
    // ISA A layout: lane M=l%16; halves = K[(l/16)*8 .. +7] then K[16+(l/16)*8 .. +7]
    const unsigned kc0b = (unsigned)((k * CC + cb * 32 + lh * 8) * 2);
    FragAB afr[4];
#pragma unroll
    for (int fm = 0; fm < 4; ++fm) {
      unsigned row = (unsigned)(m0 + wm * 64 + fm * 16 + l16);
      const char* ap = wrTb + row * (KCt * 2u) + kc0b;
      afr[fm].i2[0] = *(const v4i*)(ap);
      afr[fm].i2[1] = *(const v4i*)(ap + 32);
    }
    // ISA B layout: lane N=l%16; halves = K[(l/16)*8 .. +7] then K[16+(l/16)*8 .. +7]
    FragAB bfr[2];
#pragma unroll
    for (int fn = 0; fn < 2; ++fn) {
      int nn = wn * 32 + fn * 16 + l16;
      const _Float16* bp = &colT[p][nn * 40 + lh * 8];
      bfr[fn].i2[0] = *(const v4i*)(bp);
      bfr[fn].i2[1] = *(const v4i*)(bp + 16);
    }
#pragma unroll
    for (int fm = 0; fm < 4; ++fm)
#pragma unroll
      for (int fn = 0; fn < 2; ++fn)
        acc[fm][fn] = __builtin_amdgcn_wmma_f32_16x16x32_f16(
            false, afr[fm].h, false, bfr[fn].h, (short)0, acc[fm][fn],
            false, false);
    // no trailing barrier: next step writes the other colT buffer; overwrite of
    // this buffer happens after the *next* barrier, which follows this compute.
  }

  // ---- epilogue: D layout lane: N=l%16, M = v + 8*(l/16) ----
  const int bImg = n0 >> 12;
  const int pix0 = n0 & (HWp - 1);
  char* outb = (char*)out;
#pragma unroll
  for (int fm = 0; fm < 4; ++fm) {
    int obase = m0 + wm * 64 + fm * 16 + lh * 8;
#pragma unroll
    for (int fn = 0; fn < 2; ++fn) {
      unsigned pix = (unsigned)(pix0 + wn * 32 + fn * 16 + l16);
#pragma unroll
      for (int v = 0; v < 8; ++v) {
        int o = obase + v;
        unsigned boff = ((unsigned)(bImg * OO + o) * HWp + pix) * 4u;
        *(float*)(outb + boff) = acc[fm][fn][v] + bias[o];
      }
    }
  }
}

// ---------------------------------------------------------------------------
extern "C" void kernel_launch(void* const* d_in, const int* in_sizes, int n_in,
                              void* d_out, int out_size, void* d_ws, size_t ws_size,
                              hipStream_t stream) {
  const float* x      = (const float*)d_in[0];
  const float* offset = (const float*)d_in[1];
  const float* mask   = (const float*)d_in[2];
  const float* weight = (const float*)d_in[3];
  const float* bias   = (const float*)d_in[4];
  float* out = (float*)d_out;

  char* ws = (char*)d_ws;
  float*    xt  = (float*)ws;                                        // 16 MB
  _Float16* wrT = (_Float16*)(ws + (size_t)16 * 1024 * 1024);        // 1.125 MB (pad to 2 MB)
  float4*   sw  = (float4*)(ws + (size_t)18 * 1024 * 1024);          // 2.25 MB
  uint4*    so  = (uint4*)(ws + (size_t)18 * 1024 * 1024 +
                           (size_t)KT * NPIX * sizeof(float4));      // 2.25 MB

  k_nchw2nhwc<<<dim3(4096), dim3(256), 0, stream>>>(x, xt);
  k_wreorder<<<dim3((OO * KCt + 255) / 256), dim3(256), 0, stream>>>(weight, wrT);
  k_samp<<<dim3((KT * NPIX + 255) / 256), dim3(256), 0, stream>>>(offset, mask, sw, so);
  k_dcn_gemm<<<dim3(256), dim3(256), 0, stream>>>(wrT, xt, sw, so, bias, out);
}
